// DHModel_67783173865879
// MI455X (gfx1250) — matrix operand order repaired and checked by reading, not compiled, not run on hardware
//
#include <hip/hip_runtime.h>
#include <math.h>

// ---------------- model constants ----------------
#define IN_DIM   700
#define KP       704            // K padded to multiple of 32
#define KSTEPS   22             // KP / 32
#define HID      512
#define BRANCH   4
#define NTOT     2048           // HID * BRANCH
#define OUT_DIM  20
#define BATCH    256
#define T_STEPS  250
#define VTH      1.0f

typedef __attribute__((ext_vector_type(16))) _Float16 v16h;
typedef __attribute__((ext_vector_type(8)))  _Float16 v8h;
typedef __attribute__((ext_vector_type(8)))  float    v8f;

// ws layout sizes
#define EWP_HALFS    ((size_t)(NTOT / 16) * KSTEPS * 32 * 16)   // 1,441,792
#define EWP_BYTES    (EWP_HALFS * 2)                            // 2,883,584
#define STATE_FLOATS ((size_t)BATCH * (NTOT + HID + HID + OUT_DIM)) // 791,552

// -------------------------------------------------------------------------
// Pack (w1 * mask) -> f16 in the WMMA B-fragment layout:
// tile nt (16 N), k-step ks (32 K): lane l holds N = nt*16 + (l%16),
// K = ks*32 + (l/16)*8 + e for e<8, ks*32 + 16 + (l/16)*8 + (e-8) for e>=8. (ISA 05_wmma §7.12.2)
// -------------------------------------------------------------------------
__global__ void dh_pack_w1(const float* __restrict__ w1,
                           const float* __restrict__ mask,
                           _Float16* __restrict__ ewp) {
    const int K8 = KP / 8;                       // 88 groups of 8 K per row
    int idx = blockIdx.x * blockDim.x + threadIdx.x;
    if (idx >= NTOT * K8) return;
    int n  = idx / K8;
    int K  = (idx - n * K8) * 8;

    v8h v;
#pragma unroll
    for (int i = 0; i < 8; ++i) {
        int kk = K + i;
        float f = 0.f;
        if (kk < IN_DIM) {
            size_t off = (size_t)n * IN_DIM + kk;
            f = w1[off] * mask[off];
        }
        v[i] = (_Float16)f;
    }
    int nt = n >> 4;
    int nn = n & 15;
    int ks = K >> 5;
    int kl = K & 31;
    int g  = (kl >> 3) & 1;      // B layout: lane half selects K group of 8
    int e0 = (kl >> 4) << 3;     // 0 or 8
    int lane = nn + 16 * g;
    size_t off = ((size_t)(nt * KSTEPS + ks) * 32 + lane) * 16 + e0;
    *reinterpret_cast<v8h*>(ewp + off) = v;
}

// -------------------------------------------------------------------------
// Pack x chunk -> f16 in the WMMA A-fragment layout:
// tile mt (16 rows), k-step ks: lane l = (m%16) + 16*g holds
//   e=0..7  -> K = ks*32 + g*8 + e
//   e=8..15 -> K = ks*32 + 16 + g*8 + (e-8)
// K block of 8 (kl = K%32 in {0,8,16,24}): g = (kl>>3)&1, e0 = (kl>>4)*8.
// -------------------------------------------------------------------------
__global__ void dh_pack_x(const float* __restrict__ x,
                          _Float16* __restrict__ xp, int t0, int Tc) {
    const int K8 = KP / 8;
    int idx = blockIdx.x * blockDim.x + threadIdx.x;
    if (idx >= BATCH * Tc * K8) return;
    int r = idx / K8;                 // chunk row
    int K = (idx - r * K8) * 8;
    int b  = r / Tc;
    int tt = r - b * Tc;
    const float* __restrict__ xrow =
        x + ((size_t)b * T_STEPS + t0 + tt) * IN_DIM;

    v8h v;
#pragma unroll
    for (int i = 0; i < 8; ++i) {
        int kk = K + i;
        v[i] = (_Float16)((kk < IN_DIM) ? xrow[kk] : 0.f);
    }
    int mt = r >> 4;
    int mm = r & 15;
    int ks = K >> 5;
    int kl = K & 31;
    int g  = (kl >> 3) & 1;
    int e0 = (kl >> 4) << 3;
    int lane = mm + 16 * g;
    size_t off = ((size_t)(mt * KSTEPS + ks) * 32 + lane) * 16 + e0;
    *reinterpret_cast<v8h*>(xp + off) = v;
}

__global__ void dh_zero(float* __restrict__ p, int n) {
    int i = blockIdx.x * blockDim.x + threadIdx.x;
    if (i < n) p[i] = 0.f;
}

// -------------------------------------------------------------------------
// GEMM: cur[r, n] = sum_k x[row(r), k] * ew[n, k]  (fp32 acc via f16 WMMA)
// Pre-packed fragments; wave tile = 32M x 64N (2 A frags x 4 B frags ->
// 8 WMMAs per K-step, 24 B/lane operand traffic per WMMA).
// Block = 256 thr = 8 waves along N; block tile = 32M x 512N.
// -------------------------------------------------------------------------
__global__ void __launch_bounds__(256)
dh_gemm(const _Float16* __restrict__ xp, const _Float16* __restrict__ ewp,
        float* __restrict__ cur) {
    const int tid  = threadIdx.x;
    const int wave = tid >> 5;
    const int lane = tid & 31;
    const int mm   = lane & 15;
    const int g    = lane >> 4;
    const int m0   = blockIdx.x * 32;
    const int n0   = blockIdx.y * 512 + wave * 64;

    const size_t fragStride = (size_t)32 * 16;             // halfs per k-step
    const size_t ntStride   = (size_t)KSTEPS * fragStride; // halfs per tile
    const _Float16* __restrict__ abase =
        xp + (size_t)(blockIdx.x * 2) * ntStride + (size_t)lane * 16;
    const _Float16* __restrict__ bbase =
        ewp + (size_t)(n0 >> 4) * ntStride + (size_t)lane * 16;

    v8f acc00 = {}, acc01 = {}, acc02 = {}, acc03 = {};
    v8f acc10 = {}, acc11 = {}, acc12 = {}, acc13 = {};

    for (int ks = 0; ks < KSTEPS; ++ks) {
        const size_t o = (size_t)ks * fragStride;
        v16h a0  = *reinterpret_cast<const v16h*>(abase + o);
        v16h a1  = *reinterpret_cast<const v16h*>(abase + o + ntStride);
        v16h bf0 = *reinterpret_cast<const v16h*>(bbase + o);
        v16h bf1 = *reinterpret_cast<const v16h*>(bbase + o + ntStride);
        v16h bf2 = *reinterpret_cast<const v16h*>(bbase + o + 2 * ntStride);
        v16h bf3 = *reinterpret_cast<const v16h*>(bbase + o + 3 * ntStride);

        acc00 = __builtin_amdgcn_wmma_f32_16x16x32_f16(false, a0, false, bf0,
                                                       (short)0, acc00, false, false);
        acc01 = __builtin_amdgcn_wmma_f32_16x16x32_f16(false, a0, false, bf1,
                                                       (short)0, acc01, false, false);
        acc02 = __builtin_amdgcn_wmma_f32_16x16x32_f16(false, a0, false, bf2,
                                                       (short)0, acc02, false, false);
        acc03 = __builtin_amdgcn_wmma_f32_16x16x32_f16(false, a0, false, bf3,
                                                       (short)0, acc03, false, false);
        acc10 = __builtin_amdgcn_wmma_f32_16x16x32_f16(false, a1, false, bf0,
                                                       (short)0, acc10, false, false);
        acc11 = __builtin_amdgcn_wmma_f32_16x16x32_f16(false, a1, false, bf1,
                                                       (short)0, acc11, false, false);
        acc12 = __builtin_amdgcn_wmma_f32_16x16x32_f16(false, a1, false, bf2,
                                                       (short)0, acc12, false, false);
        acc13 = __builtin_amdgcn_wmma_f32_16x16x32_f16(false, a1, false, bf3,
                                                       (short)0, acc13, false, false);
    }

    // ---- C/D layout: lane l, vgpr v -> M = (l/16)*8 + v, N = l%16
    {
        const int rowBase = m0 + g * 8;          // A tile 0
#pragma unroll
        for (int v = 0; v < 8; ++v) {
            float* cp = cur + (size_t)(rowBase + v) * NTOT + n0 + mm;
            cp[0]  = acc00[v];
            cp[16] = acc01[v];
            cp[32] = acc02[v];
            cp[48] = acc03[v];
        }
    }
    {
        const int rowBase = m0 + 16 + g * 8;     // A tile 1
#pragma unroll
        for (int v = 0; v < 8; ++v) {
            float* cp = cur + (size_t)(rowBase + v) * NTOT + n0 + mm;
            cp[0]  = acc10[v];
            cp[16] = acc11[v];
            cp[32] = acc12[v];
            cp[48] = acc13[v];
        }
    }
}

// -------------------------------------------------------------------------
// Sequential membrane scan for one time chunk. One block per batch element,
// one hidden neuron per thread (4 branches in a float4). Readout reduced
// through LDS (320 threads x 32 MACs, float4 reads). State in workspace.
// -------------------------------------------------------------------------
__device__ __forceinline__ float dh_sigm(float v) {
    return 1.f / (1.f + expf(-v));
}

__global__ void __launch_bounds__(512)
dh_scan(const float* __restrict__ cur,
        const float* __restrict__ b1, const float* __restrict__ tau_m1,
        const float* __restrict__ tau_n1, const float* __restrict__ w2,
        const float* __restrict__ b2, const float* __restrict__ tau_m2,
        float* __restrict__ stD, float* __restrict__ stM1,
        float* __restrict__ stSpk, float* __restrict__ stM2,
        float* __restrict__ out, int t0, int Tc) {
    __shared__ float w2s[OUT_DIM * HID];
    __shared__ float spkS[HID];
    __shared__ float part[OUT_DIM * 16];

    const int tid = threadIdx.x;
    const int b   = blockIdx.x;
    const int h   = tid;

    for (int i = tid; i < OUT_DIM * HID; i += 512) w2s[i] = w2[i];

    float4 dd = *reinterpret_cast<const float4*>(stD + ((size_t)b * HID + h) * 4);
    float  m1 = stM1[b * HID + h];
    float  sp = stSpk[b * HID + h];

    float4 tn  = *reinterpret_cast<const float4*>(tau_n1 + h * 4);
    float  bt0 = dh_sigm(tn.x), bt1 = dh_sigm(tn.y);
    float  bt2 = dh_sigm(tn.z), bt3 = dh_sigm(tn.w);
    float4 bias = *reinterpret_cast<const float4*>(b1 + h * 4);
    float  alpha = dh_sigm(tau_m1[h]);

    float m2 = 0.f, a2 = 0.f, bb = 0.f;
    if (tid < OUT_DIM) {
        m2 = stM2[b * OUT_DIM + tid];
        a2 = dh_sigm(tau_m2[tid]);
        bb = b2[tid];
    }
    __syncthreads();

    for (int tt = 0; tt < Tc; ++tt) {
        const float* cr = cur + ((size_t)b * Tc + tt) * NTOT + h * 4;
        float4 c = *reinterpret_cast<const float4*>(cr);
        float c0 = c.x + bias.x, c1 = c.y + bias.y;
        float c2 = c.z + bias.z, c3 = c.w + bias.w;
        dd.x = bt0 * dd.x + (1.f - bt0) * c0;
        dd.y = bt1 * dd.y + (1.f - bt1) * c1;
        dd.z = bt2 * dd.z + (1.f - bt2) * c2;
        dd.w = bt3 * dd.w + (1.f - bt3) * c3;
        float l = dd.x + dd.y + dd.z + dd.w;
        m1 = alpha * m1 + (1.f - alpha) * l - VTH * sp;
        sp = (m1 > VTH) ? 1.f : 0.f;
        spkS[h] = sp;
        __syncthreads();

        if (tid < OUT_DIM * 16) {               // 320 threads, 32 MACs each
            int o = tid >> 4, seg = tid & 15;
            const float4* wr = reinterpret_cast<const float4*>(w2s + o * HID + seg * 32);
            const float4* sr = reinterpret_cast<const float4*>(spkS + seg * 32);
            float s = 0.f;
#pragma unroll
            for (int i = 0; i < 8; ++i) {
                float4 wv = wr[i];
                float4 sv = sr[i];
                s += sv.x * wv.x + sv.y * wv.y + sv.z * wv.z + sv.w * wv.w;
            }
            part[tid] = s;
        }
        __syncthreads();

        if (tid < OUT_DIM) {
            float s = bb;
#pragma unroll
            for (int i = 0; i < 16; ++i) s += part[tid * 16 + i];
            m2 = a2 * m2 + (1.f - a2) * s;
            out[((size_t)b * T_STEPS + t0 + tt) * OUT_DIM + tid] = m2;
        }
        // next part write happens after the next top barrier (which part
        // readers also cross) -> two barriers per step suffice
    }

    __syncthreads();
    *reinterpret_cast<float4*>(stD + ((size_t)b * HID + h) * 4) = dd;
    stM1[b * HID + h]  = m1;
    stSpk[b * HID + h] = sp;
    if (tid < OUT_DIM) stM2[b * OUT_DIM + tid] = m2;
}

// -------------------------------------------------------------------------
extern "C" void kernel_launch(void* const* d_in, const int* in_sizes, int n_in,
                              void* d_out, int out_size, void* d_ws, size_t ws_size,
                              hipStream_t stream) {
    const float* x      = (const float*)d_in[0];
    const float* w1     = (const float*)d_in[1];
    const float* b1     = (const float*)d_in[2];
    const float* tau_m1 = (const float*)d_in[3];
    const float* tau_n1 = (const float*)d_in[4];
    const float* mask   = (const float*)d_in[5];
    const float* w2     = (const float*)d_in[6];
    const float* b2     = (const float*)d_in[7];
    const float* tau_m2 = (const float*)d_in[8];
    float* out = (float*)d_out;

    char* ws = (char*)d_ws;
    _Float16* ewp = (_Float16*)ws;
    float* stD   = (float*)(ws + EWP_BYTES);
    float* stM1  = stD  + (size_t)BATCH * NTOT;
    float* stSpk = stM1 + (size_t)BATCH * HID;
    float* stM2  = stSpk + (size_t)BATCH * HID;
    // after fixed region: [xpack f16: B*Tc*KP][cur f32: B*Tc*NTOT]
    char* dynBase = (char*)(stM2 + (size_t)BATCH * OUT_DIM);

    // pick the largest time-chunk (divisor of 250) that fits in workspace
    size_t fixedBytes = EWP_BYTES + STATE_FLOATS * 4;
    size_t avail = (ws_size > fixedBytes) ? (ws_size - fixedBytes) : 0;
    size_t perT  = (size_t)BATCH * (NTOT * 4 + KP * 2);   // cur + xpack per t
    long tcmax = (long)(avail / perT);
    const int cand[8] = {250, 125, 50, 25, 10, 5, 2, 1};
    int Tc = 1;
    for (int i = 0; i < 8; ++i) {
        if ((long)cand[i] <= tcmax) { Tc = cand[i]; break; }
    }

    _Float16* xpack = (_Float16*)dynBase;
    float* curbuf = (float*)(dynBase + (size_t)BATCH * Tc * KP * 2);

    // one-time packs / state init
    int packN = NTOT * (KP / 8);
    dh_pack_w1<<<(packN + 255) / 256, 256, 0, stream>>>(w1, mask, ewp);
    int zn = (int)STATE_FLOATS;
    dh_zero<<<(zn + 255) / 256, 256, 0, stream>>>(stD, zn);

    for (int t0 = 0; t0 < T_STEPS; t0 += Tc) {
        int pxN = BATCH * Tc * (KP / 8);
        dh_pack_x<<<(pxN + 255) / 256, 256, 0, stream>>>(x, xpack, t0, Tc);
        dim3 ggrid(BATCH * Tc / 32, NTOT / 512);
        dh_gemm<<<ggrid, 256, 0, stream>>>(xpack, ewp, curbuf);
        dh_scan<<<BATCH, 512, 0, stream>>>(curbuf, b1, tau_m1, tau_n1, w2, b2,
                                           tau_m2, stD, stM1, stSpk, stM2,
                                           out, t0, Tc);
    }
}
